// CustomDiceLoss_55207509623392
// MI455X (gfx1250) — compile-verified, matrix operand order checked by reading, not run on hardware
//
#include <hip/hip_runtime.h>
#include <hip/hip_bf16.h>

typedef _Float16 v16h __attribute__((ext_vector_type(16)));
typedef float    v8f  __attribute__((ext_vector_type(8)));

#define HW    9216
#define WDIM  96
#define TILE  16
#define NTILE (HW / TILE)          // 576 tiles of 16 pixels
#define WAVES_PER_BLOCK 4
#define ITILES_PER_WAVE 2          // two 16-row tiles per wave share one B-frag
#define PEN   16384.0f             // same-class penalty; 2*PEN=32768 > max d2 (18050); exact in f16

// ---------------------------------------------------------------------------
// Kernel 1: border weights via WMMA pairwise-distance tiles.
// A row i carries (h_i, w_i, -P*c_i, -P*(1-c_i)); B col j carries
// (h_j, w_j, c_j, 1-c_j). D = A*B gives cross = h.h + w.w - P*[same class],
// so sq_i + sq_j - 2*D is the exact squared distance, +2P when same class.
// Track min_j (sq_j - 2*D) per row, add sq_i at the end. All values are
// integers exactly representable in f16 inputs / f32 accumulation.
// Branchless fragment builds: lane predicate applied as masks, so the
// 576-iteration loop has no exec-mask save/restore.
// ---------------------------------------------------------------------------
__global__ __launch_bounds__(128)
void border_weights_kernel(const int* __restrict__ targets,
                           float* __restrict__ weights)
{
    __shared__ unsigned s_cls[HW];     // packed (half c, half 1-c) per pixel

    const int b    = blockIdx.y;
    const int tid  = threadIdx.x;
    const int lane = tid & 31;
    const int wave = tid >> 5;
    const int* t   = targets + b * HW;

    __builtin_prefetch(t, 0, 0);       // global_prefetch_b8

    // Stage class bits into LDS as packed (c, 1-c) halves
    for (int idx = tid; idx < HW; idx += 128) {
        float c = (float)t[idx];
        _Float16 h0 = (_Float16)c;
        _Float16 h1 = (_Float16)(1.0f - c);
        unsigned u = (unsigned)__builtin_bit_cast(unsigned short, h0) |
                     ((unsigned)__builtin_bit_cast(unsigned short, h1) << 16);
        s_cls[idx] = u;
    }
    __syncthreads();

    const int ibase0 = (blockIdx.x * WAVES_PER_BLOCK + wave) * (ITILES_PER_WAVE * TILE);
    const int ibase1 = ibase0 + TILE;

    // Loop-invariant lane predicates (branchless masking of upper half-wave)
    const float    fm = (lane < 16) ? 1.0f : 0.0f;
    const unsigned lm = (lane < 16) ? 0xffffffffu : 0u;
    const int      col = lane & 15;

    // ---- A fragments for the two row tiles (16-bit A 16x32 layout:
    //      lanes 0-15 row M=lane hold K=0..7 in elems 0..7; rest zero)
    v16h afrag0 = {}, afrag1 = {};
    {
        int r0 = ibase0 + col, r1 = ibase1 + col;
        int h0 = r0 / WDIM, w0 = r0 - h0 * WDIM;
        int h1 = r1 / WDIM, w1 = r1 - h1 * WDIM;
        unsigned cu0 = s_cls[r0] & lm;
        unsigned cu1 = s_cls[r1] & lm;
        _Float16 negP = (_Float16)(-PEN);
        afrag0[0] = (_Float16)((float)h0 * fm);
        afrag0[1] = (_Float16)((float)w0 * fm);
        afrag0[2] = negP * __builtin_bit_cast(_Float16, (unsigned short)(cu0 & 0xffffu));
        afrag0[3] = negP * __builtin_bit_cast(_Float16, (unsigned short)(cu0 >> 16));
        afrag1[0] = (_Float16)((float)h1 * fm);
        afrag1[1] = (_Float16)((float)w1 * fm);
        afrag1[2] = negP * __builtin_bit_cast(_Float16, (unsigned short)(cu1 & 0xffffu));
        afrag1[3] = negP * __builtin_bit_cast(_Float16, (unsigned short)(cu1 >> 16));
    }

    // Running min of (sq_j - 2*cross) per C/D slot, one set per row tile
    v8f accmin0, accmin1;
    #pragma unroll
    for (int r = 0; r < 8; ++r) { accmin0[r] = 3.0e38f; accmin1[r] = 3.0e38f; }

    int hj = 0, wj = col;              // column pixel coords; advance by 16/iter

    for (int jbase = 0; jbase < HW; jbase += TILE) {
        float fh = (float)hj, fw = (float)wj;
        float sqj = fh * fh + fw * fw;

        // B fragment (32x16, K-major): lanes 0-15 hold K=0..7,16..23 for col N=lane,
        // lanes 16-31 hold K=8..15,24..31 (masked to zero).
        unsigned cu = s_cls[jbase + col] & lm;
        v16h bfrag = {};
        bfrag[0] = (_Float16)(fh * fm);
        bfrag[1] = (_Float16)(fw * fm);
        bfrag[2] = __builtin_bit_cast(_Float16, (unsigned short)(cu & 0xffffu));
        bfrag[3] = __builtin_bit_cast(_Float16, (unsigned short)(cu >> 16));

        v8f c0 = {};
        v8f d0 = __builtin_amdgcn_wmma_f32_16x16x32_f16(
            false, afrag0, false, bfrag, (short)0, c0, false, false);
        v8f d1 = __builtin_amdgcn_wmma_f32_16x16x32_f16(
            false, afrag1, false, bfrag, (short)0, c0, false, false);

        #pragma unroll
        for (int r = 0; r < 8; ++r) {
            accmin0[r] = fminf(accmin0[r], __builtin_fmaf(-2.0f, d0[r], sqj));
            accmin1[r] = fminf(accmin1[r], __builtin_fmaf(-2.0f, d1[r], sqj));
        }

        wj += TILE;
        if (wj >= WDIM) { wj -= WDIM; hj += 1; }
    }

    // Horizontal min across the 16-lane half groups (M=r in lanes 0-15,
    // M=r+8 in lanes 16-31; xor offsets < 16 stay within each half)
    #pragma unroll
    for (int r = 0; r < 8; ++r) {
        float v0 = accmin0[r], v1 = accmin1[r];
        v0 = fminf(v0, __shfl_xor(v0, 8, 32));
        v1 = fminf(v1, __shfl_xor(v1, 8, 32));
        v0 = fminf(v0, __shfl_xor(v0, 4, 32));
        v1 = fminf(v1, __shfl_xor(v1, 4, 32));
        v0 = fminf(v0, __shfl_xor(v0, 2, 32));
        v1 = fminf(v1, __shfl_xor(v1, 2, 32));
        v0 = fminf(v0, __shfl_xor(v0, 1, 32));
        v1 = fminf(v1, __shfl_xor(v1, 1, 32));
        accmin0[r] = v0; accmin1[r] = v1;
    }

    if ((lane & 15) == 0) {            // lane 0 -> rows 0..7, lane 16 -> rows 8..15
        int rbase = (lane >> 4) * 8;
        #pragma unroll
        for (int r = 0; r < 8; ++r) {
            int row0 = ibase0 + rbase + r;
            int row1 = ibase1 + rbase + r;
            int h0 = row0 / WDIM, w0 = row0 - h0 * WDIM;
            int h1 = row1 / WDIM, w1 = row1 - h1 * WDIM;
            float d20 = fmaxf((float)(h0 * h0 + w0 * w0) + accmin0[r], 0.0f);
            float d21 = fmaxf((float)(h1 * h1 + w1 * w1) + accmin1[r], 0.0f);
            // 10 * exp(-dmin / (2*sigma^2)) with sigma=5 -> /50
            weights[b * HW + row0] = 10.0f * __expf(-sqrtf(d20) * 0.02f);
            weights[b * HW + row1] = 10.0f * __expf(-sqrtf(d21) * 0.02f);
        }
    }
}

// ---------------------------------------------------------------------------
// Kernel 2: deterministic per-batch dice partial sums (one block per batch).
// acc[b*8+0..4] = { sum w*p*t, sum w*(p+t), sum p*t, sum (p+t), sum t }
// ---------------------------------------------------------------------------
__global__ __launch_bounds__(256)
void dice_reduce_kernel(const float* __restrict__ p,
                        const int*   __restrict__ t,
                        const float* __restrict__ w,
                        float* __restrict__ acc)
{
    const int b = blockIdx.x;
    const float* pb = p + b * HW;
    const int*   tb = t + b * HW;
    const float* wb = w + b * HW;

    float s0 = 0.f, s1 = 0.f, s2 = 0.f, s3 = 0.f, s4 = 0.f;
    for (int idx = threadIdx.x; idx < HW; idx += 256) {
        float pv = pb[idx];
        float tv = (float)tb[idx];
        float wv = wb[idx];
        float pt = pv * tv;
        float su = pv + tv;
        s0 += wv * pt;  s1 += wv * su;  s2 += pt;  s3 += su;  s4 += tv;
    }

    #pragma unroll
    for (int off = 16; off; off >>= 1) {
        s0 += __shfl_xor(s0, off, 32);
        s1 += __shfl_xor(s1, off, 32);
        s2 += __shfl_xor(s2, off, 32);
        s3 += __shfl_xor(s3, off, 32);
        s4 += __shfl_xor(s4, off, 32);
    }

    __shared__ float red[8][5];
    if ((threadIdx.x & 31) == 0) {
        int wv = threadIdx.x >> 5;
        red[wv][0] = s0; red[wv][1] = s1; red[wv][2] = s2;
        red[wv][3] = s3; red[wv][4] = s4;
    }
    __syncthreads();
    if (threadIdx.x == 0) {
        float tot[5] = {0.f, 0.f, 0.f, 0.f, 0.f};
        for (int wv = 0; wv < 8; ++wv)
            for (int k = 0; k < 5; ++k) tot[k] += red[wv][k];
        for (int k = 0; k < 5; ++k) acc[b * 8 + k] = tot[k];
    }
}

// ---------------------------------------------------------------------------
// Kernel 3: final scalar loss = sum_b (1 - num/den), with the mask.sum()<=1
// all-ones-weights fallback from the reference.
// ---------------------------------------------------------------------------
__global__ void dice_final_kernel(const float* __restrict__ acc,
                                  float* __restrict__ out, int B)
{
    if (threadIdx.x == 0 && blockIdx.x == 0) {
        float loss = 0.0f;
        for (int b = 0; b < B; ++b) {
            const float* a = acc + b * 8;
            float num, den;
            if (a[4] > 1.0f) { num = 2.0f * a[0] + 1.0f; den = a[1] + 1.0f; }
            else             { num = 2.0f * a[2] + 1.0f; den = a[3] + 1.0f; }
            loss += 1.0f - num / den;
        }
        out[0] = loss;
    }
}

// ---------------------------------------------------------------------------
extern "C" void kernel_launch(void* const* d_in, const int* in_sizes, int n_in,
                              void* d_out, int out_size, void* d_ws, size_t ws_size,
                              hipStream_t stream)
{
    const float* inputs  = (const float*)d_in[0];   // [B,1,96,96] f32
    const int*   targets = (const int*)d_in[1];     // [B,1,96,96] i32
    float*       out     = (float*)d_out;           // scalar loss
    float*       wsf     = (float*)d_ws;

    const int B = in_sizes[0] / HW;

    float* weights = wsf;            // [B*HW]
    float* acc     = wsf + B * HW;   // [B*8]

    // 576 itiles / (4 waves * 2 itiles per wave) = 72 blocks per batch
    dim3 g1(NTILE / (WAVES_PER_BLOCK * ITILES_PER_WAVE), B);
    border_weights_kernel<<<g1, 128, 0, stream>>>(targets, weights);

    dice_reduce_kernel<<<dim3(B), 256, 0, stream>>>(inputs, targets, weights, acc);

    dice_final_kernel<<<1, 32, 0, stream>>>(acc, out, B);
}
